// TFAttentionModule_14620068676282
// MI455X (gfx1250) — matrix operand order repaired and checked by reading, not compiled, not run on hardware
//
#include <hip/hip_runtime.h>

// TF dual-attention (time + freq) for MI455X / gfx1250.
// Pipeline (per attention, run twice reusing workspace):
//   1) conv1x1 -> q,k written bf16 in transposed [row][(c,*)] layout
//   2) conv1x1 -> v written bf16 (transposed for time, natural for freq)
//   3) energy GEMM-NT (bf16 WMMA, K=4096) -> f32
//   4) row softmax -> bf16 attn
//   5) out GEMM-NT (bf16 WMMA, K=512) fused epilogue: out = 3x + g_t*Ot (+= g_f*Of)
// GEMM waves use 64x64 register tiles: 16 WMMAs per 8 fragment loads
// (32 FLOP/byte from cache) to keep v_wmma fed from L0/L2.
// Workspace needed: ~166 MB.

#define NN 4
#define CH 64
#define C8V 8
#define HH 512
#define WW 512
#define KFEAT 4096   // C8 * 512

typedef __attribute__((ext_vector_type(16))) __bf16 v16bf;
typedef __attribute__((ext_vector_type(8)))  float  v8f;

union BFFrag { v16bf v; uint4 q[2]; };

__device__ inline unsigned short f2bf(float f) {
    unsigned u = __float_as_uint(f);
    u += 0x7FFFu + ((u >> 16) & 1u);     // round to nearest even
    return (unsigned short)(u >> 16);
}

// ---- WMMA fragment loads (bf16, 16x16x32; layouts per CDNA5 ISA 7.12.2) ----
// A (16x32 MxK): lane L<16 holds row L, K={0..7,16..23}; lane 16+L holds K={8..15,24..31}
__device__ inline v16bf load_a_frag(const unsigned short* __restrict__ A, int lda,
                                    int m0, int k0, int lane) {
    const int row = m0 + (lane & 15);
    const int hi  = (lane >> 4) ? 8 : 0;
    const unsigned short* p = A + (size_t)row * lda + k0 + hi;
    BFFrag f;
    f.q[0] = *(const uint4*)(p);
    f.q[1] = *(const uint4*)(p + 16);
    return f.v;
}
// B (32x16 KxN), stored as row-major Bt[n][k]: lane L<16 = col L, K={0..7,16..23}; lane 16+L = col L, K={8..15,24..31}
__device__ inline v16bf load_b_frag(const unsigned short* __restrict__ B, int ldb,
                                    int n0, int k0, int lane) {
    const int col  = n0 + (lane & 15);
    const int koff = (lane >> 4) ? 8 : 0;
    const unsigned short* p = B + (size_t)col * ldb + k0 + koff;
    BFFrag f;
    f.q[0] = *(const uint4*)(p);
    f.q[1] = *(const uint4*)(p + 16);
    return f.v;
}

__device__ inline v8f wmma_bf16(v16bf a, v16bf b, v8f c) {
    return __builtin_amdgcn_wmma_f32_16x16x32_bf16(false, a, false, b, (short)0, c,
                                                   false, false);
}

// ---------------- Stage 1a: q,k conv1x1 (64 -> 8 each), transposed writes ----
__global__ __launch_bounds__(128) void qk_conv_kernel(
    const float* __restrict__ x,
    const float* __restrict__ wq, const float* __restrict__ bq,
    const float* __restrict__ wk, const float* __restrict__ bk,
    unsigned short* __restrict__ qout, unsigned short* __restrict__ kout,
    int timeMode)
{
    __shared__ float xs[CH][128];
    __shared__ float wqs[C8V * CH], wks[C8V * CH];
    __shared__ float bqs[C8V], bks[C8V];
    const int tid = threadIdx.x;
    const int w0  = blockIdx.x * 128;
    const int h   = blockIdx.y;
    const int n   = blockIdx.z;
    for (int i = tid; i < C8V * CH; i += 128) { wqs[i] = wq[i]; wks[i] = wk[i]; }
    if (tid < C8V) { bqs[tid] = bq[tid]; bks[tid] = bk[tid]; }
    for (int c = 0; c < CH; ++c)
        xs[c][tid] = x[(((size_t)n * CH + c) * HH + h) * WW + w0 + tid];
    __syncthreads();

    float qa[C8V], ka[C8V];
#pragma unroll
    for (int o = 0; o < C8V; ++o) { qa[o] = bqs[o]; ka[o] = bks[o]; }
    for (int c = 0; c < CH; ++c) {
        const float xv = xs[c][tid];
#pragma unroll
        for (int o = 0; o < C8V; ++o) {
            qa[o] = fmaf(wqs[o * CH + c], xv, qa[o]);
            ka[o] = fmaf(wks[o * CH + c], xv, ka[o]);
        }
    }
    const int w = w0 + tid;
#pragma unroll
    for (int o = 0; o < C8V; ++o) {
        size_t idx;
        if (timeMode) idx = ((size_t)n * HH + h) * KFEAT + o * WW + w;  // q_t[n][h][c*W+w]
        else          idx = ((size_t)n * WW + w) * KFEAT + o * HH + h;  // q_f[n][w][c*H+h]
        qout[idx] = f2bf(qa[o]);
        kout[idx] = f2bf(ka[o]);
    }
}

// ---------------- Stage 1b: v conv1x1 (64 -> 64) --------------------------
__global__ __launch_bounds__(128) void v_conv_kernel(
    const float* __restrict__ x,
    const float* __restrict__ wv, const float* __restrict__ bv,
    unsigned short* __restrict__ vout, int timeMode)
{
    __shared__ float xs[CH][128];
    __shared__ float wvs[CH * CH];
    const int tid = threadIdx.x;
    const int w0  = blockIdx.x * 128;
    const int h   = blockIdx.y;
    const int n   = blockIdx.z;
    for (int i = tid; i < CH * CH; i += 128) wvs[i] = wv[i];
    for (int c = 0; c < CH; ++c)
        xs[c][tid] = x[(((size_t)n * CH + c) * HH + h) * WW + w0 + tid];
    __syncthreads();

    float acc[CH];
#pragma unroll
    for (int o = 0; o < CH; ++o) acc[o] = bv[o];
    for (int c = 0; c < CH; ++c) {
        const float xv = xs[c][tid];
#pragma unroll
        for (int o = 0; o < CH; ++o) acc[o] = fmaf(wvs[o * CH + c], xv, acc[o]);
    }
    const int w = w0 + tid;
#pragma unroll
    for (int o = 0; o < CH; ++o) {
        size_t idx;
        if (timeMode) idx = (((size_t)n * CH + o) * WW + w) * HH + h;  // vT[n][c][w][g=h]
        else          idx = (((size_t)n * CH + o) * HH + h) * WW + w;  // v_f natural
        vout[idx] = f2bf(acc[o]);
    }
}

// ---- 64x64-per-wave GEMM-NT core: C[64x64] += A[64xK] * B[64xK]^T --------
// acc[i][j] is the 16x16 subtile at rows m0+16i, cols n0+16j.
__device__ inline void gemm64x64(const unsigned short* __restrict__ A,
                                 const unsigned short* __restrict__ B,
                                 int ld, int Kdim, int m0, int n0, int lane,
                                 v8f acc[4][4])
{
    for (int k0 = 0; k0 < Kdim; k0 += 32) {
        v16bf fa[4], fb[4];
#pragma unroll
        for (int i = 0; i < 4; ++i) fa[i] = load_a_frag(A, ld, m0 + 16 * i, k0, lane);
#pragma unroll
        for (int j = 0; j < 4; ++j) fb[j] = load_b_frag(B, ld, n0 + 16 * j, k0, lane);
#pragma unroll
        for (int i = 0; i < 4; ++i)
#pragma unroll
            for (int j = 0; j < 4; ++j)
                acc[i][j] = wmma_bf16(fa[i], fb[j], acc[i][j]);
    }
}

// ---------------- Energy GEMM-NT: E[h][g] = Q[h][:] . K[g][:], K=4096 -----
__global__ __launch_bounds__(128) void energy_kernel(
    const unsigned short* __restrict__ Q, const unsigned short* __restrict__ Km,
    float* __restrict__ E)
{
    const int lane = threadIdx.x & 31;
    const int wid  = threadIdx.x >> 5;
    const int n    = blockIdx.y;
    const int tile = blockIdx.x * 4 + wid;          // 64 tiles of 64x64 per n
    const int m0   = (tile >> 3) * 64;
    const int n0   = (tile & 7) * 64;
    const unsigned short* A = Q  + (size_t)n * HH * KFEAT;
    const unsigned short* B = Km + (size_t)n * HH * KFEAT;

    v8f acc[4][4] = {};
    gemm64x64(A, B, KFEAT, KFEAT, m0, n0, lane, acc);

    float* Erow = E + (size_t)n * HH * HH;
    const int colb = (lane & 15);
    const int rowb = (lane >> 4) << 3;
#pragma unroll
    for (int i = 0; i < 4; ++i) {
        const int rbase = m0 + 16 * i + rowb;
#pragma unroll
        for (int j = 0; j < 4; ++j) {
            const int col = n0 + 16 * j + colb;
#pragma unroll
            for (int r = 0; r < 8; ++r)
                Erow[(size_t)(rbase + r) * HH + col] = acc[i][j][r];
        }
    }
}

// ---------------- Softmax over last axis of [2048][512], f32 -> bf16 ------
__global__ __launch_bounds__(256) void softmax_kernel(
    const float* __restrict__ E, unsigned short* __restrict__ Aout)
{
    const int lane = threadIdx.x & 31;
    const int wid  = threadIdx.x >> 5;
    const int row  = blockIdx.x * 8 + wid;          // 2048 rows total
    const float* e = E + (size_t)row * 512;
    float vals[16];
    float m = -3.402823466e38f;
#pragma unroll
    for (int j = 0; j < 16; ++j) { vals[j] = e[lane + j * 32]; m = fmaxf(m, vals[j]); }
    for (int off = 16; off; off >>= 1) m = fmaxf(m, __shfl_xor(m, off, 32));
    float s = 0.f;
#pragma unroll
    for (int j = 0; j < 16; ++j) { vals[j] = __expf(vals[j] - m); s += vals[j]; }
    for (int off = 16; off; off >>= 1) s += __shfl_xor(s, off, 32);
    const float inv = 1.f / s;
    unsigned short* arow = Aout + (size_t)row * 512;
#pragma unroll
    for (int j = 0; j < 16; ++j) arow[lane + j * 32] = f2bf(vals[j] * inv);
}

// ---------------- Out GEMM-NT, K=512, fused residual epilogue -------------
// accumulateMode==0: out = 3*x + g*acc   (time pass, writes)
// accumulateMode==1: out += g*acc        (freq pass, accumulates)
__global__ __launch_bounds__(128) void out_kernel(
    const unsigned short* __restrict__ Abase, const unsigned short* __restrict__ Bbase,
    const float* __restrict__ x, const float* __restrict__ gamma,
    float* __restrict__ out, int attnIsA, int accumulateMode)
{
    const int lane = threadIdx.x & 31;
    const int wid  = threadIdx.x >> 5;
    const int c    = blockIdx.y;
    const int n    = blockIdx.z;
    const int tile = blockIdx.x * 4 + wid;          // 64 tiles of 64x64 per (n,c)
    const int m0   = (tile >> 3) * 64;
    const int n0   = (tile & 7) * 64;
    // time: A = attn_t[n] (512x512), B = vT[n][c] (512x512)
    // freq: A = v_f[n][c],           B = attn_f[n]
    const unsigned short* A = attnIsA ? Abase + (size_t)n * 512 * 512
                                      : Abase + ((size_t)n * CH + c) * 512 * 512;
    const unsigned short* B = attnIsA ? Bbase + ((size_t)n * CH + c) * 512 * 512
                                      : Bbase + (size_t)n * 512 * 512;

    v8f acc[4][4] = {};
    gemm64x64(A, B, 512, 512, m0, n0, lane, acc);

    const float g = gamma[0];
    float* O       = out + ((size_t)n * CH + c) * HH * WW;
    const float* X = x   + ((size_t)n * CH + c) * HH * WW;
    const int colb = (lane & 15);
    const int rowb = (lane >> 4) << 3;
#pragma unroll
    for (int i = 0; i < 4; ++i) {
        const int rbase = m0 + 16 * i + rowb;
#pragma unroll
        for (int j = 0; j < 4; ++j) {
            const int col = n0 + 16 * j + colb;
#pragma unroll
            for (int r = 0; r < 8; ++r) {
                const size_t idx = (size_t)(rbase + r) * WW + col;
                if (accumulateMode) O[idx] += g * acc[i][j][r];
                else                O[idx] = 3.f * X[idx] + g * acc[i][j][r];
            }
        }
    }
}

extern "C" void kernel_launch(void* const* d_in, const int* in_sizes, int n_in,
                              void* d_out, int out_size, void* d_ws, size_t ws_size,
                              hipStream_t stream) {
    const float* x    = (const float*)d_in[0];
    const float* tq_w = (const float*)d_in[1];
    const float* tq_b = (const float*)d_in[2];
    const float* tk_w = (const float*)d_in[3];
    const float* tk_b = (const float*)d_in[4];
    const float* tv_w = (const float*)d_in[5];
    const float* tv_b = (const float*)d_in[6];
    const float* t_g  = (const float*)d_in[7];
    const float* fq_w = (const float*)d_in[8];
    const float* fq_b = (const float*)d_in[9];
    const float* fk_w = (const float*)d_in[10];
    const float* fk_b = (const float*)d_in[11];
    const float* fv_w = (const float*)d_in[12];
    const float* fv_b = (const float*)d_in[13];
    const float* f_g  = (const float*)d_in[14];
    float* out = (float*)d_out;

    // Workspace layout (bytes), total ~166 MB, reused across time/freq phases.
    char* ws = (char*)d_ws;
    unsigned short* qb = (unsigned short*)(ws);                      // 16 MB
    unsigned short* kb = (unsigned short*)(ws + (16ull  << 20));     // 16 MB
    unsigned short* vb = (unsigned short*)(ws + (32ull  << 20));     // 128 MB
    float*          eb = (float*)         (ws + (160ull << 20));     // 4 MB
    unsigned short* ab = (unsigned short*)(ws + (164ull << 20));     // 2 MB

    const dim3 convGrid(WW / 128, HH, NN), convBlk(128);
    const dim3 eGrid(16, NN), gemmBlk(128);
    const dim3 smGrid(256), smBlk(256);
    const dim3 oGrid(16, CH, NN);

    // ---- time attention ----
    qk_conv_kernel<<<convGrid, convBlk, 0, stream>>>(x, tq_w, tq_b, tk_w, tk_b, qb, kb, 1);
    v_conv_kernel<<<convGrid, convBlk, 0, stream>>>(x, tv_w, tv_b, vb, 1);
    energy_kernel<<<eGrid, gemmBlk, 0, stream>>>(qb, kb, eb);
    softmax_kernel<<<smGrid, smBlk, 0, stream>>>(eb, ab);
    out_kernel<<<oGrid, gemmBlk, 0, stream>>>(ab, vb, x, t_g, out, /*attnIsA=*/1, /*accum=*/0);

    // ---- freq attention ----
    qk_conv_kernel<<<convGrid, convBlk, 0, stream>>>(x, fq_w, fq_b, fk_w, fk_b, qb, kb, 0);
    v_conv_kernel<<<convGrid, convBlk, 0, stream>>>(x, fv_w, fv_b, vb, 0);
    energy_kernel<<<eGrid, gemmBlk, 0, stream>>>(qb, kb, eb);
    softmax_kernel<<<smGrid, smBlk, 0, stream>>>(eb, ab);
    out_kernel<<<oGrid, gemmBlk, 0, stream>>>(vb, ab, x, f_g, out, /*attnIsA=*/0, /*accum=*/1);
}
